// Self_Attention_8083128451440
// MI455X (gfx1250) — matrix operand order repaired and checked by reading, not compiled, not run on hardware
//
#include <hip/hip_runtime.h>
#include <math.h>

// Problem constants (B, S, D, H) = (2, 2048, 1024, 16), HD = 64
#define BB 2
#define SS 2048
#define DDIM 1024
#define HH 16
#define HDIM 64

typedef __bf16 bf16_t;
typedef __bf16 v16bf __attribute__((ext_vector_type(16)));
typedef __bf16 v8bf  __attribute__((ext_vector_type(8)));
typedef float  v8f   __attribute__((ext_vector_type(8)));
typedef unsigned int v4u __attribute__((ext_vector_type(4)));
typedef int v8i __attribute__((ext_vector_type(8)));
typedef int v4i __attribute__((ext_vector_type(4)));

__device__ __forceinline__ bf16_t f32_to_bf16(float f) {
  unsigned u = __builtin_bit_cast(unsigned, f);
  unsigned r = (u + 0x7FFFu + ((u >> 16) & 1u)) >> 16;
  return __builtin_bit_cast(bf16_t, (unsigned short)r);
}

__device__ __forceinline__ v16bf cat8(v8bf lo, v8bf hi) {
  return __builtin_shufflevector(lo, hi, 0, 1, 2, 3, 4, 5, 6, 7,
                                         8, 9, 10, 11, 12, 13, 14, 15);
}

__device__ __forceinline__ v8f wmma_bf16(v16bf a, v16bf b, v8f c) {
  // D = A(16x32 bf16) * B(32x16 bf16) + C(16x16 f32)
  return __builtin_amdgcn_wmma_f32_16x16x32_bf16(
      /*neg_a=*/false, a, /*neg_b=*/false, b,
      /*c_mod=*/(short)0, c, /*reuse_a=*/false, /*reuse_b=*/false);
}

// Per-lane async global->LDS 16B copy (tracked by ASYNCcnt).
__device__ __forceinline__ void async_b128(void* lds, const void* gmem) {
  unsigned ldsOff = (unsigned)(size_t)lds;           // generic LDS ptr: low 32b = LDS offset
  unsigned long long ga = (unsigned long long)gmem;
  asm volatile("global_load_async_to_lds_b128 %0, %1, off"
               :: "v"(ldsOff), "v"(ga)
               : "memory");
}

__device__ __forceinline__ void wait_asynccnt0() {
  asm volatile("s_wait_asynccnt 0" ::: "memory");
}

// TDM: load a 2-D bf16 tile (tileX x tileY elements, row stride strideElems)
// from global memory into contiguous LDS at ldsDst.  One instruction per wave.
__device__ __forceinline__ void tdm_load_2d_bf16(void* ldsDst, const void* gsrc,
                                                 unsigned tileX, unsigned tileY,
                                                 unsigned strideElems) {
  unsigned long long ga = (unsigned long long)gsrc;
  v4u g0;
  g0.x = 1u;                                             // count=1, user D#
  g0.y = (unsigned)(size_t)ldsDst;                       // lds_addr (bytes)
  g0.z = (unsigned)ga;                                   // global_addr[31:0]
  g0.w = (unsigned)((ga >> 32) & 0x01FFFFFFu) | (2u << 30);  // addr[56:32] | type=2
  v8i g1;
  g1[0] = (int)(1u << 16);          // workgroup_mask=0, data_size=1 (2 bytes)
  g1[1] = (int)(tileX << 16);       // tensor_dim0[15:0] at bit48
  g1[2] = (int)(tileY << 16);       // tensor_dim0[31:16]=0, tensor_dim1[15:0] at bit80
  g1[3] = (int)(tileX << 16);       // tensor_dim1 hi=0, tile_dim0 at bit112
  g1[4] = (int)tileY;               // tile_dim1 at bit128, tile_dim2=0
  g1[5] = (int)strideElems;         // tensor_dim0_stride[31:0]
  g1[6] = 0;                        // stride hi, tensor_dim1_stride lo
  g1[7] = 0;
  v4i gz4 = {};
  v8i gz8 = {};
  // 6-arg toolchain variant: (g0, g1, g2, g3, <extra group regs>, cpol)
  __builtin_amdgcn_tensor_load_to_lds(g0, g1, gz4, gz4, gz8, 0);
}

// ---------------------------------------------------------------------------
// Generic tiled GEMM:  C[M,N] = A[M,K] @ B[K,N] + bias[N]
// A: f32 or bf16 row-major; B: f32 row-major (converted to bf16 on the fly);
// C: f32 or bf16 row-major.  Block = 128 threads (4 wave32), tile 64x64, K-step 32.
// Optionally scatters the V third of the qkv output into a pre-transposed
// V_t[b][h][hd][s] buffer for the flash kernel's async V staging.
// ---------------------------------------------------------------------------
template <bool A_F32, bool OUT_F32>
__global__ __launch_bounds__(128) void gemm_bias_wmma(
    const void* __restrict__ Ap, const float* __restrict__ Bw,
    const float* __restrict__ bias, void* __restrict__ Cp,
    bf16_t* __restrict__ vT,
    int M, int N, int K, int lda, int ldb, int ldc) {
  __shared__ __align__(16) bf16_t Als[64][40];  // [m][k]   (pad: 80B row = 16B mult)
  __shared__ __align__(16) bf16_t Bls[64][40];  // [n][k]   (transposed for B-operand)

  const int tid  = threadIdx.x;
  const int lane = tid & 31;
  const int wave = tid >> 5;
  const int l15  = lane & 15;
  const int hi   = lane >> 4;
  const int m0   = blockIdx.y * 64;
  const int n0   = blockIdx.x * 64;

  v8f acc[4] = {};  // wave computes 16 rows x 64 cols (4 chunks of 16)

  for (int k0 = 0; k0 < K; k0 += 32) {
    // ---- stage A tile (64 x 32) into LDS as bf16, coalesced ----
#pragma unroll
    for (int i = 0; i < 16; ++i) {
      int idx = tid + i * 128;           // 2048 elements
      int r = idx >> 5, c = idx & 31;
      if (A_F32) {
        float v = ((const float*)Ap)[(size_t)(m0 + r) * lda + (k0 + c)];
        Als[r][c] = f32_to_bf16(v);
      } else {
        Als[r][c] = ((const bf16_t*)Ap)[(size_t)(m0 + r) * lda + (k0 + c)];
      }
    }
    // ---- stage B tile (32 x 64) transposed into LDS: Bls[n][kk] ----
#pragma unroll
    for (int i = 0; i < 16; ++i) {
      int idx = tid + i * 128;           // 2048 elements
      int n = idx & 63, kk = idx >> 6;   // global reads coalesced along n
      Bls[n][kk] = f32_to_bf16(Bw[(size_t)(k0 + kk) * ldb + (n0 + n)]);
    }
    // ---- prefetch next K-step tiles (global_prefetch_b8) ----
    if (k0 + 32 < K) {
      size_t aElem = (size_t)(m0 + (tid >> 1)) * lda + (k0 + 32) + (tid & 1) * 16;
      __builtin_prefetch((const char*)Ap + aElem * (A_F32 ? 4 : 2), 0, 3);
      size_t bElem = (size_t)(k0 + 32 + (tid >> 2)) * ldb + n0 + (tid & 3) * 16;
      __builtin_prefetch((const char*)Bw + bElem * 4, 0, 3);
    }
    __syncthreads();

    // ---- A operand: lane row = wave*16 + l15; K = hi*8+{0..7}, 16+hi*8+{0..7}
    const int arow = wave * 16 + l15;
    v16bf aOp = cat8(*(const v8bf*)&Als[arow][hi * 8],
                     *(const v8bf*)&Als[arow][16 + hi * 8]);
#pragma unroll
    for (int c = 0; c < 4; ++c) {
      // B operand: lane col = c*16 + l15; K = hi*8+{0..7}, 16+hi*8+{0..7}
      const int bcol = c * 16 + l15;
      v16bf bOp = cat8(*(const v8bf*)&Bls[bcol][hi * 8],
                       *(const v8bf*)&Bls[bcol][16 + hi * 8]);
      acc[c] = wmma_bf16(aOp, bOp, acc[c]);
    }
    __syncthreads();
  }

  // ---- epilogue: C layout lane holds (row = r + 8*hi, col = c*16 + l15) ----
#pragma unroll
  for (int c = 0; c < 4; ++c) {
#pragma unroll
    for (int r = 0; r < 8; ++r) {
      int row = m0 + wave * 16 + r + 8 * hi;
      int col = n0 + c * 16 + l15;
      float v = acc[c][r] + bias[col];
      if (OUT_F32) {
        ((float*)Cp)[(size_t)row * ldc + col] = v;
      } else {
        ((bf16_t*)Cp)[(size_t)row * ldc + col] = f32_to_bf16(v);
        // Extra pre-transposed copy of V for the flash kernel's async staging.
        if (vT != nullptr && col >= 2 * DDIM) {
          int h  = (col - 2 * DDIM) >> 6;
          int hd = (col - 2 * DDIM) & 63;
          int bb = row >> 11;                 // row = b*S + s, S = 2048
          int s  = row & (SS - 1);
          vT[(((size_t)bb * HH + h) * HDIM + hd) * SS + s] = f32_to_bf16(v);
        }
      }
    }
  }
}

// ---------------------------------------------------------------------------
// Flash attention (causal), bf16 WMMA, f32 accumulate / online softmax.
// qkv layout (from GEMM1): [b, s, 3, h, hd] bf16.  vT: [b, h, hd, s] bf16.
// Output: [b, s, h, hd] bf16.
// Block: 128 threads = 4 waves; block handles (b,h) and 64 q rows (16 per wave).
// K tile staged by the Tensor Data Mover (TENSORcnt); V tile staged with
// per-lane async global->LDS b128 copies (ASYNCcnt).
// ---------------------------------------------------------------------------
__global__ __launch_bounds__(128) void flash_attn(
    const bf16_t* __restrict__ qkv, const bf16_t* __restrict__ vT,
    bf16_t* __restrict__ o_out) {
  const int bh = blockIdx.y;
  const int b  = bh / HH;
  const int h  = bh % HH;
  const int q0 = blockIdx.x * 64;

  const int tid  = threadIdx.x;
  const int lane = tid & 31;
  const int wave = tid >> 5;
  const int l15  = lane & 15;
  const int hi   = lane >> 4;

  __shared__ __align__(16) bf16_t Kls[32][64];        // [key][hd]   (TDM fill)
  __shared__ __align__(16) bf16_t Vls[64][32];        // [hd][key]   (async fill)
  __shared__ __align__(16) bf16_t Pls[4][16][32];     // per-wave P scratch

  const float SCALE = 0.125f;   // 1/sqrt(64)
  const float NEG   = -1.0e30f;

  // ---- load Q operands once (A layout, rows = q0 + wave*16 + l15) ----
  const int qrow = q0 + wave * 16 + l15;
  const bf16_t* qbase =
      qkv + ((size_t)b * SS + qrow) * (3 * DDIM) + (size_t)h * HDIM;
  v16bf qa0 = cat8(*(const v8bf*)(qbase + hi * 8),
                   *(const v8bf*)(qbase + 16 + hi * 8));        // hd 0..31
  v16bf qa1 = cat8(*(const v8bf*)(qbase + 32 + hi * 8),
                   *(const v8bf*)(qbase + 48 + hi * 8));        // hd 32..63

  float m_run[8], l_run[8];
#pragma unroll
  for (int r = 0; r < 8; ++r) { m_run[r] = NEG; l_run[r] = 0.f; }
  v8f oacc[4] = {};

  const bf16_t* vtBase = vT + ((size_t)b * HH + h) * HDIM * SS;  // [hd][s]

  const int nT = (q0 + 64) / 32;   // uniform over the block (causal coverage)
  for (int t = 0; t < nT; ++t) {
    // ---- K tile [32 keys][64 hd] via Tensor Data Mover (wave 0 issues) ----
    if (wave == 0) {
      const bf16_t* kg =
          qkv + ((size_t)b * SS + t * 32) * (3 * DDIM) + DDIM + (size_t)h * HDIM;
      tdm_load_2d_bf16(&Kls[0][0], kg, /*tileX=*/HDIM, /*tileY=*/32,
                       /*strideElems=*/3 * DDIM);
    }
    // ---- V tile [hd][key] via async global->LDS (all waves, 16B/lane) ----
#pragma unroll
    for (int i = 0; i < 2; ++i) {
      int idx = tid + i * 128;                  // 256 x 16B chunks
      int hd = idx >> 2, ch = idx & 3;
      async_b128(&Vls[hd][ch * 8], vtBase + (size_t)hd * SS + t * 32 + ch * 8);
    }
    __builtin_amdgcn_s_wait_tensorcnt(0);
    wait_asynccnt0();
    __syncthreads();

    // ---- logits = Q @ K^T for 32 keys (two 16-key groups, chained over hd) ----
    v8f lg[2];
#pragma unroll
    for (int g = 0; g < 2; ++g) {
      const int kcol = g * 16 + l15;
      v16bf kbLo = cat8(*(const v8bf*)&Kls[kcol][hi * 8],
                        *(const v8bf*)&Kls[kcol][16 + hi * 8]);       // hd 0..31
      v16bf kbHi = cat8(*(const v8bf*)&Kls[kcol][32 + hi * 8],
                        *(const v8bf*)&Kls[kcol][48 + hi * 8]);  // hd 32..63
      v8f z = {};
      z = wmma_bf16(qa0, kbLo, z);
      lg[g] = wmma_bf16(qa1, kbHi, z);
    }

    // ---- scale + causal mask; online softmax statistics ----
    float s0[8], s1[8];
#pragma unroll
    for (int r = 0; r < 8; ++r) {
      int qabs = q0 + wave * 16 + r + 8 * hi;
      int k0a  = t * 32 + l15;
      int k1a  = t * 32 + 16 + l15;
      s0[r] = (k0a <= qabs) ? lg[0][r] * SCALE : NEG;
      s1[r] = (k1a <= qabs) ? lg[1][r] * SCALE : NEG;
    }
    float alpha[8], newm[8];
#pragma unroll
    for (int r = 0; r < 8; ++r) {
      float mx = fmaxf(s0[r], s1[r]);
      mx = fmaxf(mx, __shfl_xor(mx, 1));
      mx = fmaxf(mx, __shfl_xor(mx, 2));
      mx = fmaxf(mx, __shfl_xor(mx, 4));
      mx = fmaxf(mx, __shfl_xor(mx, 8));        // reduce over 16-lane half
      newm[r]  = fmaxf(m_run[r], mx);
      alpha[r] = __expf(m_run[r] - newm[r]);
      m_run[r] = newm[r];
    }
    float p0[8], p1[8];
#pragma unroll
    for (int r = 0; r < 8; ++r) {
      p0[r] = __expf(s0[r] - newm[r]);
      p1[r] = __expf(s1[r] - newm[r]);
      float rs = p0[r] + p1[r];
      rs += __shfl_xor(rs, 1);
      rs += __shfl_xor(rs, 2);
      rs += __shfl_xor(rs, 4);
      rs += __shfl_xor(rs, 8);
      l_run[r] = l_run[r] * alpha[r] + rs;
    }
#pragma unroll
    for (int c = 0; c < 4; ++c)
#pragma unroll
      for (int r = 0; r < 8; ++r) oacc[c][r] *= alpha[r];

    // ---- P (C layout) -> LDS -> A layout ----
#pragma unroll
    for (int r = 0; r < 8; ++r) {
      Pls[wave][r + 8 * hi][l15]      = f32_to_bf16(p0[r]);
      Pls[wave][r + 8 * hi][16 + l15] = f32_to_bf16(p1[r]);
    }
    __syncthreads();
    v16bf pa = cat8(*(const v8bf*)&Pls[wave][l15][hi * 8],
                    *(const v8bf*)&Pls[wave][l15][16 + hi * 8]);

    // ---- O += P(16x32) @ V(32x64), 4 N-chunks of 16 ----
#pragma unroll
    for (int c = 0; c < 4; ++c) {
      const int vcol = c * 16 + l15;
      v16bf vb = cat8(*(const v8bf*)&Vls[vcol][hi * 8],
                      *(const v8bf*)&Vls[vcol][16 + hi * 8]);
      oacc[c] = wmma_bf16(pa, vb, oacc[c]);
    }
    __syncthreads();   // protect Kls/Vls/Pls before next iteration restages
  }

  // ---- normalize and store as bf16 [b, s, h, hd] ----
#pragma unroll
  for (int r = 0; r < 8; ++r) {
    float inv = 1.0f / l_run[r];
    int row = q0 + wave * 16 + r + 8 * hi;
#pragma unroll
    for (int c = 0; c < 4; ++c) {
      int col = h * HDIM + c * 16 + l15;
      o_out[((size_t)b * SS + row) * DDIM + col] = f32_to_bf16(oacc[c][r] * inv);
    }
  }
}

// ---------------------------------------------------------------------------
extern "C" void kernel_launch(void* const* d_in, const int* in_sizes, int n_in,
                              void* d_out, int out_size, void* d_ws, size_t ws_size,
                              hipStream_t stream) {
  (void)in_sizes; (void)n_in; (void)out_size; (void)ws_size;
  const float* x     = (const float*)d_in[0];   // [B,S,D]
  const float* qkv_w = (const float*)d_in[1];   // [D, 3D]
  const float* qkv_b = (const float*)d_in[2];   // [3D]
  const float* out_w = (const float*)d_in[3];   // [D, D]
  const float* out_b = (const float*)d_in[4];   // [D]
  float* out = (float*)d_out;                   // [B,S,D]

  bf16_t* qkv_ws  = (bf16_t*)d_ws;                            // [B,S,3,H,HD] bf16
  bf16_t* attn_ws = qkv_ws + (size_t)BB * SS * 3 * DDIM;      // [B,S,H,HD]  bf16
  bf16_t* vT_ws   = attn_ws + (size_t)BB * SS * DDIM;         // [B,H,HD,S]  bf16

  const int M = BB * SS;     // 4096
  dim3 blk(128);

  // 1) QKV projection: [M, D] @ [D, 3D] + b  -> bf16 workspace (+ transposed V)
  gemm_bias_wmma<true, false><<<dim3((3 * DDIM) / 64, M / 64), blk, 0, stream>>>(
      x, qkv_w, qkv_b, qkv_ws, vT_ws, M, 3 * DDIM, DDIM, DDIM, 3 * DDIM, 3 * DDIM);

  // 2) Causal flash attention per (b, h), 64 q-rows per block
  flash_attn<<<dim3(SS / 64, BB * HH), blk, 0, stream>>>(qkv_ws, vT_ws, attn_ws);

  // 3) Output projection: [M, D] @ [D, D] + b -> f32 output
  gemm_bias_wmma<false, true><<<dim3(DDIM / 64, M / 64), blk, 0, stream>>>(
      attn_ws, out_w, out_b, out, nullptr, M, DDIM, DDIM, DDIM, DDIM, DDIM);
}
